// Biaffine_57475252355702
// MI455X (gfx1250) — matrix operand order repaired and checked by reading, not compiled, not run on hardware
//
#include <hip/hip_runtime.h>
#include <hip/hip_bf16.h>

#define DD 1024
#define BB 8
#define PP 64
#define SS 512
#define LL 64

typedef __attribute__((ext_vector_type(16))) __bf16 v16bf;
typedef __attribute__((ext_vector_type(8)))  __bf16 v8bf;
typedef __attribute__((ext_vector_type(8)))  float  v8f;

__device__ __forceinline__ unsigned short f2bf_bits(float f) {
    union { float f; unsigned u; } x; x.f = f;
    unsigned r = x.u + 0x7FFFu + ((x.u >> 16) & 1u);   // round-to-nearest-even
    return (unsigned short)(r >> 16);
}
__device__ __forceinline__ float bf2f(unsigned short h) {
    union { unsigned u; float f; } x; x.u = ((unsigned)h) << 16;
    return x.f;
}

// LDS byte offset (relative to wave LDS base) of a __shared__ pointer.
__device__ __forceinline__ unsigned lds_off(const void* p) {
    return (unsigned)(unsigned long long)(
        (__attribute__((address_space(3))) const char*)p);
}

// gfx1250 async copy: 16 bytes global -> LDS per lane, tracked by ASYNCcnt.
__device__ __forceinline__ void async_copy_b128(unsigned ldsDst, const unsigned short* gsrc) {
    asm volatile("global_load_async_to_lds_b128 %0, %1, off"
                 :: "v"(ldsDst), "v"(gsrc) : "memory");
}
__device__ __forceinline__ void wait_asynccnt0() {
    asm volatile("s_wait_asynccnt 0x0" ::: "memory");
}

// ---------------- prep kernels ----------------

__global__ void cvt_bf16_kernel(const float* __restrict__ src,
                                unsigned short* __restrict__ dst, size_t n) {
    size_t i = (size_t)blockIdx.x * blockDim.x + threadIdx.x;
    size_t stride = (size_t)gridDim.x * blockDim.x;
    for (; i < n; i += stride) dst[i] = f2bf_bits(src[i]);
}

__global__ void gather_pred_kernel(const float* __restrict__ span,
                                   const int* __restrict__ preds,
                                   unsigned short* __restrict__ predB) {
    int bp = blockIdx.x;              // b*PP + p
    int b  = bp >> 6;
    int s  = preds[bp];
    const float* src = span + ((size_t)b * SS + s) * DD;
    unsigned short* dst = predB + (size_t)bp * DD;
    for (int d = threadIdx.x; d < DD; d += 256) dst[d] = f2bf_bits(src[d]);
}

// W1T[l][e][d] = W1[l][d][e], f32 -> bf16
__global__ void transpose_cvt_kernel(const float* __restrict__ src,
                                     unsigned short* __restrict__ dst) {
    __shared__ unsigned short tile[32][33];
    int l = blockIdx.z;
    const float* S = src + (size_t)l * DD * DD;
    unsigned short* T = dst + (size_t)l * DD * DD;
    int e_in = blockIdx.x * 32 + threadIdx.x;
    for (int i = 0; i < 32; i += 8) {
        int d = blockIdx.y * 32 + threadIdx.y + i;
        tile[threadIdx.y + i][threadIdx.x] = f2bf_bits(S[(size_t)d * DD + e_in]);
    }
    __syncthreads();
    int d_out = blockIdx.y * 32 + threadIdx.x;
    for (int i = 0; i < 32; i += 8) {
        int e = blockIdx.x * 32 + threadIdx.y + i;
        T[(size_t)e * DD + d_out] = tile[threadIdx.x][threadIdx.y + i];
    }
}

// ---------------- WMMA GEMM ----------------
// C[z][m][n] = sum_k A[m][k] * Bcm[n][k]  (A row-major lda=K; Bcm = B^T
// row-major ldb=K so each B column has K contiguous).
// Block = 256 thr = 8 waves. Block tile 64(M) x 256(N); wave tile 64x32
// (4 M-tiles x 2 N-tiles, 8 f32 accumulators).
// A slab (64x32 bf16) is staged Global->LDS with async-copy, double-buffered.
#define A_ROWSTRIDE 40            // shorts: 80B row stride, 16B aligned, bank-spread
#define A_BUFELEMS  (64 * A_ROWSTRIDE)

template <bool F32OUT>
__global__ __launch_bounds__(256)
void gemm_bf16_kernel(const unsigned short* __restrict__ A,
                      const unsigned short* __restrict__ Bm,
                      void* __restrict__ Cout,
                      int N, int K,
                      long long aHi, long long aLo,
                      long long bHi, long long bLo,
                      long long cStr, int loCount) {
    __shared__ unsigned short shA[2][A_BUFELEMS];

    int z  = blockIdx.z;
    int hi = z / loCount, lo = z % loCount;
    const unsigned short* Ab = A  + (size_t)hi * aHi + (size_t)lo * aLo;
    const unsigned short* Bb = Bm + (size_t)hi * bHi + (size_t)lo * bLo;

    int wave = threadIdx.x >> 5;
    int lane = threadIdx.x & 31;
    int m0 = blockIdx.y * 64;
    int n0 = blockIdx.x * 256 + wave * 32;

    int l15   = lane & 15;        // A row within tile / B column within tile
    int khalf = lane >> 4;        // lane-half selects K sub-range
    // B k order chosen here (same as A): lanes0-15 K0-15, lanes16-31 K16-31 (contig per lane)
    const unsigned short* bPtr0 = Bb + (size_t)(n0 + l15) * K + khalf * 16;
    const unsigned short* bPtr1 = Bb + (size_t)(n0 + 16 + l15) * K + khalf * 16;

    // async-copy mapping: each of 256 threads moves 16B of the 64x32 A slab
    int cpRow   = threadIdx.x >> 2;          // 0..63
    int cpChunk = threadIdx.x & 3;           // 4 x 16B per row
    const unsigned short* gsrc = Ab + (size_t)(m0 + cpRow) * K + cpChunk * 8;
    unsigned ldsBase = lds_off(&shA[0][0]);
    unsigned myDst   = (unsigned)(cpRow * A_ROWSTRIDE + cpChunk * 8) * 2u; // bytes
    const unsigned bufBytes = A_BUFELEMS * 2u;

    // prologue: stage k=0 slab into buffer 0
    async_copy_b128(ldsBase + myDst, gsrc);
    wait_asynccnt0();
    __syncthreads();

    v8f acc[4][2] = {};
    int cur = 0;

    for (int k = 0; k < K; k += 32) {
        if (k + 32 < K)
            async_copy_b128(ldsBase + (cur ^ 1) * bufBytes + myDst, gsrc + k + 32);

        v16bf bf0 = *reinterpret_cast<const v16bf*>(bPtr0 + k);
        v16bf bf1 = *reinterpret_cast<const v16bf*>(bPtr1 + k);

        const unsigned short* sA = &shA[cur][0];
#pragma unroll
        for (int mt = 0; mt < 4; ++mt) {
            // A k order chosen here (same as B): lanes0-15 K 0..15,
            //                        lanes16-31 K 16..31 (contig per lane)
            const unsigned short* ap = sA + (mt * 16 + l15) * A_ROWSTRIDE + khalf * 16;
            v8bf alo = *reinterpret_cast<const v8bf*>(ap);
            v8bf ahi = *reinterpret_cast<const v8bf*>(ap + 8);
            v16bf afrag = __builtin_shufflevector(alo, ahi,
                0, 1, 2, 3, 4, 5, 6, 7, 8, 9, 10, 11, 12, 13, 14, 15);
            acc[mt][0] = __builtin_amdgcn_wmma_f32_16x16x32_bf16(
                false, afrag, false, bf0, (short)0, acc[mt][0], false, false);
            acc[mt][1] = __builtin_amdgcn_wmma_f32_16x16x32_bf16(
                false, afrag, false, bf1, (short)0, acc[mt][1], false, false);
        }

        wait_asynccnt0();          // my prefetch landed
        __syncthreads();           // everyone's landed; everyone done reading cur
        cur ^= 1;
    }

    // C 16x16 f32 layout: VGPR v holds M=v (lanes0-15) / M=v+8 (lanes16-31), N=lane&15
    int rbase = khalf * 8;
#pragma unroll
    for (int mt = 0; mt < 4; ++mt) {
#pragma unroll
        for (int nt = 0; nt < 2; ++nt) {
            int ccol = n0 + nt * 16 + l15;
#pragma unroll
            for (int v = 0; v < 8; ++v) {
                size_t idx = (size_t)z * cStr
                           + (size_t)(m0 + mt * 16 + rbase + v) * N + ccol;
                if (F32OUT) ((float*)Cout)[idx] = acc[mt][nt][v];
                else        ((unsigned short*)Cout)[idx] = f2bf_bits(acc[mt][nt][v]);
            }
        }
    }
}

// ---------------- small f32 kernels ----------------

// out[r] = sum_j relu(H[r][j] + bias[j]) * w[j]
__global__ void score_reduce_kernel(const unsigned short* __restrict__ H,
                                    const float* __restrict__ bias,
                                    const float* __restrict__ w,
                                    float* __restrict__ out) {
    __shared__ float sdata[256];
    size_t r = blockIdx.x;
    float acc = 0.f;
    for (int j = threadIdx.x; j < DD; j += 256) {
        float h = bf2f(H[r * DD + j]) + bias[j];
        h = h > 0.f ? h : 0.f;
        acc += h * w[j];
    }
    sdata[threadIdx.x] = acc;
    __syncthreads();
    for (int s = 128; s > 0; s >>= 1) {
        if (threadIdx.x < s) sdata[threadIdx.x] += sdata[threadIdx.x + s];
        __syncthreads();
    }
    if (threadIdx.x == 0) out[r] = sdata[0];
}

// out[r][l] = sum_d x[d] * W2p[d][l]   (x gathered through preds when given)
__global__ void second_kernel(const float* __restrict__ span,
                              const int* __restrict__ preds,
                              const float* __restrict__ W2p,
                              float* __restrict__ out,
                              int rowsPerBatch) {
    __shared__ float xs[DD];
    __shared__ float partial[4][LL];
    int r = blockIdx.x;
    const float* x;
    if (preds) {
        int b = r / rowsPerBatch;
        x = span + ((size_t)b * SS + preds[r]) * DD;
    } else {
        x = span + (size_t)r * DD;
    }
    for (int i = threadIdx.x; i < DD; i += 256) xs[i] = x[i];
    __syncthreads();
    int l = threadIdx.x & 63, q = threadIdx.x >> 6;
    float acc = 0.f;
    for (int d = q * 256; d < (q + 1) * 256; ++d) acc += xs[d] * W2p[d * LL + l];
    partial[q][l] = acc;
    __syncthreads();
    if (q == 0)
        out[(size_t)r * LL + l] = partial[0][l] + partial[1][l] + partial[2][l] + partial[3][l];
}

// scores[((b*P+p)*S+s)*L + l] = first[b][l][p][s] + sp + sa + bias + p_score + a_score
__global__ void epilogue_kernel(const float* __restrict__ first,
                                const float* __restrict__ sp,
                                const float* __restrict__ sa,
                                const float* __restrict__ biasL,
                                const float* __restrict__ pscore,
                                const float* __restrict__ ascore,
                                float* __restrict__ scores) {
    int l  = threadIdx.x & 63;
    int ds = threadIdx.x >> 6;
    int s  = blockIdx.x * 4 + ds;
    int p  = blockIdx.y;
    int b  = blockIdx.z;
    size_t row = ((size_t)b * PP + p) * SS + s;
    float v = first[(((size_t)b * LL + l) * PP + p) * SS + s]
            + sp[((size_t)b * PP + p) * LL + l]
            + sa[((size_t)b * SS + s) * LL + l]
            + biasL[l]
            + pscore[b * PP + p] + ascore[b * SS + s];
    if (l == LL - 1) v = 0.f;
    scores[row * LL + l] = v;
}

__global__ void labels_kernel(const int* __restrict__ labels,
                              float* __restrict__ out, int n) {
    int i = blockIdx.x * 256 + threadIdx.x;
    if (i < n) out[i] = (float)labels[i];
}

// ---------------- host ----------------

extern "C" void kernel_launch(void* const* d_in, const int* in_sizes, int n_in,
                              void* d_out, int out_size, void* d_ws, size_t ws_size,
                              hipStream_t stream) {
    const float* span  = (const float*)d_in[0];
    const int*   preds = (const int*)d_in[1];
    const int*   labels= (const int*)d_in[2];
    const float* Wp    = (const float*)d_in[4];   // [2][D][D]
    const float* bp    = (const float*)d_in[5];   // [2][D]
    const float* Wa    = (const float*)d_in[6];
    const float* ba    = (const float*)d_in[7];
    const float* w_p   = (const float*)d_in[8];
    const float* w_a   = (const float*)d_in[9];
    const float* W1    = (const float*)d_in[10];  // [L][D][D]
    const float* W2    = (const float*)d_in[11];  // [2D][L]
    const float* bb    = (const float*)d_in[12];  // [L]
    float* out = (float*)d_out;

    char* base = (char*)d_ws;
    size_t off = 0;
    auto carve = [&](size_t bytes) -> void* {
        void* p = base + off;
        off += (bytes + 255) & ~(size_t)255;
        return p;
    };
    unsigned short* spanB = (unsigned short*)carve((size_t)BB*SS*DD*2);
    unsigned short* predB = (unsigned short*)carve((size_t)BB*PP*DD*2);
    unsigned short* W1T   = (unsigned short*)carve((size_t)LL*DD*DD*2);
    unsigned short* WaB   = (unsigned short*)carve((size_t)DD*DD*2);
    unsigned short* WpB   = (unsigned short*)carve((size_t)DD*DD*2);
    unsigned short* U     = (unsigned short*)carve((size_t)BB*LL*PP*DD*2);
    unsigned short* Ha    = (unsigned short*)carve((size_t)BB*SS*DD*2);
    unsigned short* Hp    = (unsigned short*)carve((size_t)BB*PP*DD*2);
    float* first  = (float*)carve((size_t)BB*LL*PP*SS*4);
    float* ascore = (float*)carve((size_t)BB*SS*4);
    float* pscore = (float*)carve((size_t)BB*PP*4);
    float* sa     = (float*)carve((size_t)BB*SS*LL*4);
    float* sp     = (float*)carve((size_t)BB*PP*LL*4);

    // 1) conversions / gathers / transpose
    cvt_bf16_kernel<<<2048, 256, 0, stream>>>(span, spanB, (size_t)BB*SS*DD);
    gather_pred_kernel<<<BB*PP, 256, 0, stream>>>(span, preds, predB);
    transpose_cvt_kernel<<<dim3(DD/32, DD/32, LL), dim3(32, 8), 0, stream>>>(W1, W1T);
    cvt_bf16_kernel<<<1024, 256, 0, stream>>>(Wa + (size_t)DD*DD, WaB, (size_t)DD*DD);
    cvt_bf16_kernel<<<1024, 256, 0, stream>>>(Wp + (size_t)DD*DD, WpB, (size_t)DD*DD);

    // 2) GEMM1: U[b][l] = pred[b] @ W1[l]   (M=64, N=1024, K=1024, 512 batches)
    gemm_bf16_kernel<false><<<dim3(DD/256, 1, BB*LL), 256, 0, stream>>>(
        predB, W1T, U, DD, DD,
        (long long)PP*DD, 0LL, 0LL, (long long)DD*DD,
        (long long)PP*DD, LL);

    // 3) GEMM2: first[b][l] = U[b][l] @ span[b]^T  (M=64, N=512, K=1024)
    gemm_bf16_kernel<true><<<dim3(SS/256, 1, BB*LL), 256, 0, stream>>>(
        U, spanB, first, SS, DD,
        (long long)LL*PP*DD, (long long)PP*DD,
        (long long)SS*DD, 0LL,
        (long long)PP*SS, LL);

    // 4) GEMM3: Ha = span @ Wa1^T (M=4096), Hp = pred @ Wp1^T (M=512)
    gemm_bf16_kernel<false><<<dim3(DD/256, (BB*SS)/64, 1), 256, 0, stream>>>(
        spanB, WaB, Ha, DD, DD, 0LL, 0LL, 0LL, 0LL, 0LL, 1);
    gemm_bf16_kernel<false><<<dim3(DD/256, (BB*PP)/64, 1), 256, 0, stream>>>(
        predB, WpB, Hp, DD, DD, 0LL, 0LL, 0LL, 0LL, 0LL, 1);

    // 5) unary scores
    score_reduce_kernel<<<BB*SS, 256, 0, stream>>>(Ha, ba + DD, w_a, ascore);
    score_reduce_kernel<<<BB*PP, 256, 0, stream>>>(Hp, bp + DD, w_p, pscore);

    // 6) second terms
    second_kernel<<<BB*SS, 256, 0, stream>>>(span, nullptr, W2 + (size_t)DD*LL, sa, SS);
    second_kernel<<<BB*PP, 256, 0, stream>>>(span, preds, W2, sp, PP);

    // 7) combine + labels tail
    epilogue_kernel<<<dim3(SS/4, PP, BB), 256, 0, stream>>>(
        first, sp, sa, bb, pscore, ascore, out);
    labels_kernel<<<(BB*PP*SS)/256, 256, 0, stream>>>(
        labels, out + (size_t)BB*PP*SS*LL, BB*PP*SS);
}
